// AttentionLayer_15676630630786
// MI455X (gfx1250) — compile-verified, matrix operand order checked by reading, not run on hardware
//
#include <hip/hip_runtime.h>
#include <stdint.h>

typedef __attribute__((ext_vector_type(16))) __bf16 v16bf;
typedef __attribute__((ext_vector_type(8)))  __bf16 v8bf;
typedef __attribute__((ext_vector_type(4)))  __bf16 v4bf;
typedef __attribute__((ext_vector_type(8)))  float  v8f;
typedef __attribute__((ext_vector_type(4)))  float  v4f;

#define N_  4
#define P_  2048
#define D_  512
#define H_  8
#define HD_ 64

__device__ __forceinline__ __bf16 f2bf(float f) { return (__bf16)f; }

__device__ __forceinline__ v16bf load_ab_bf16(const __bf16* __restrict__ src, int ld) {
  const int lane = threadIdx.x & 31;
  const __bf16* p = src + (size_t)(lane & 15) * ld + ((lane >> 4) << 3);
  v8bf lo = *(const v8bf*)p;
  v8bf hi = *(const v8bf*)(p + 16);
  v16bf a;
#pragma unroll
  for (int i = 0; i < 8; ++i) { a[i] = lo[i]; a[i + 8] = hi[i]; }
  return a;
}

__device__ __forceinline__ v16bf load_ab_f32(const float* __restrict__ src, int ld) {
  const int lane = threadIdx.x & 31;
  const float* p = src + (size_t)(lane & 15) * ld + ((lane >> 4) << 3);
  v4f c0 = *(const v4f*)(p);
  v4f c1 = *(const v4f*)(p + 4);
  v4f c2 = *(const v4f*)(p + 16);
  v4f c3 = *(const v4f*)(p + 20);
  v16bf a;
#pragma unroll
  for (int i = 0; i < 4; ++i) {
    a[i] = f2bf(c0[i]); a[i + 4]  = f2bf(c1[i]);
    a[i + 8] = f2bf(c2[i]); a[i + 12] = f2bf(c3[i]);
  }
  return a;
}

__device__ __forceinline__ v8f wmma_bf16(v16bf a, v16bf b, v8f c) {
  return __builtin_amdgcn_wmma_f32_16x16x32_bf16(false, a, false, b, (short)0, c,
                                                 false, false);
}

__global__ void cvt_wts_kernel(const float* __restrict__ Wq,
                               const float* __restrict__ Wk,
                               const float* __restrict__ Wv,
                               const float* __restrict__ Wo,
                               __bf16* __restrict__ Wqkv_bf,
                               __bf16* __restrict__ Wo_bf) {
  int g = blockIdx.x * blockDim.x + threadIdx.x;
  const float* src;
  __bf16* dst;
  int off;
  if (g < (D_ * D_) / 4) {
    src = Wo; dst = Wo_bf; off = g * 4;
  } else {
    int g2 = g - (D_ * D_) / 4;
    if (g2 >= 3 * (HD_ * HD_) / 4) return;
    int w = g2 >> 10;
    off = (g2 & 1023) * 4;
    src = (w == 0) ? Wq : (w == 1) ? Wk : Wv;
    dst = Wqkv_bf + w * HD_ * HD_;
  }
  v4f v = *(const v4f*)(src + off);
  v4bf b;
#pragma unroll
  for (int q = 0; q < 4; ++q) b[q] = f2bf(v[q]);
  *(v4bf*)(dst + off) = b;
}

__global__ void qkv_kernel(const float* __restrict__ x,
                           const __bf16* __restrict__ Wqkv_bf,
                           __bf16* __restrict__ Q, __bf16* __restrict__ K,
                           __bf16* __restrict__ Vt) {
  const int wave = threadIdx.x >> 5;
  const int lane = threadIdx.x & 31;
  const int gw = blockIdx.x * 8 + wave;
  const int nh = gw >> 7;
  const int p0 = (gw & 127) * 16;
  const int n = nh >> 3;
  const int h = nh & (H_ - 1);
  const int nc = lane & 15;
  const int mb = (lane >> 4) << 3;

  const float* xb = x + ((size_t)(n * P_ + p0) * H_ + h) * HD_;   // ld = 512
  v16bf a0 = load_ab_f32(xb, D_);
  v16bf a1 = load_ab_f32(xb + 32, D_);

#pragma unroll
  for (int w = 0; w < 3; ++w) {
    const __bf16* W = Wqkv_bf + w * HD_ * HD_;
#pragma unroll
    for (int e0 = 0; e0 < HD_; e0 += 16) {
      v16bf b0 = load_ab_bf16(W + e0 * HD_, HD_);
      v16bf b1 = load_ab_bf16(W + e0 * HD_ + 32, HD_);
      v8f c = {};
      c = wmma_bf16(a0, b0, c);
      c = wmma_bf16(a1, b1, c);
      if (w < 2) {
        __bf16* dst = (w == 0) ? Q : K;
#pragma unroll
        for (int j = 0; j < 8; ++j) {
          int p = p0 + mb + j;
          dst[((size_t)(n * P_ + p) * H_ + h) * HD_ + e0 + nc] = f2bf(c[j]);
        }
      } else {
        v8bf vv;
#pragma unroll
        for (int j = 0; j < 8; ++j) vv[j] = f2bf(c[j]);
        *(v8bf*)&Vt[((size_t)nh * HD_ + e0 + nc) * P_ + p0 + mb] = vv;
      }
    }
  }
}

__global__ void attn_kernel(const __bf16* __restrict__ Q,
                            const __bf16* __restrict__ K,
                            const __bf16* __restrict__ Vt,
                            float* __restrict__ attn_out,
                            __bf16* __restrict__ Oh) {
  extern __shared__ char smem_raw[];
  float*  E     = (float*)smem_raw;
  __bf16* Abf   = (__bf16*)(smem_raw + 16 * P_ * 4);
  float*  red   = (float*)(smem_raw + 16 * P_ * 4 + 16 * P_ * 2);
  float*  rstat = red + 128;

  const int wave = threadIdx.x >> 5;
  const int lane = threadIdx.x & 31;
  const int nh = blockIdx.x >> 7;
  const int rb = blockIdx.x & 127;
  const int n = nh >> 3;
  const int h = nh & (H_ - 1);
  const int p0 = rb * 16;
  const int nc = lane & 15;
  const int mb = (lane >> 4) << 3;
  const float scale = 0.04419417382415922f;  // 1/sqrt(512)

  const __bf16* Qb = Q + ((size_t)(n * P_ + p0) * H_ + h) * HD_;
  const __bf16* Kb = K + ((size_t)n * P_ * H_ + h) * HD_;

  v16bf a0 = load_ab_bf16(Qb, D_);
  v16bf a1 = load_ab_bf16(Qb + 32, D_);

  for (int kt = wave; kt < P_ / 16; kt += 4) {
    const __bf16* Kt = Kb + (size_t)kt * 16 * D_;
    if (kt + 4 < P_ / 16)
      __builtin_prefetch(Kb + (size_t)(kt + 4) * 16 * D_, 0, 1);
    v16bf b0 = load_ab_bf16(Kt, D_);
    v16bf b1 = load_ab_bf16(Kt + 32, D_);
    v8f c = {};
    c = wmma_bf16(a0, b0, c);
    c = wmma_bf16(a1, b1, c);
#pragma unroll
    for (int j = 0; j < 8; ++j)
      E[(size_t)(mb + j) * P_ + kt * 16 + nc] = c[j] * scale;
  }
  __syncthreads();

  const int t = threadIdx.x;
  const int r = t >> 3;
  const int i0 = (t & 7) * 256;
  float* Er = E + (size_t)r * P_ + i0;

  float m = -3.4e38f;
  for (int i = 0; i < 256; i += 4) {
    v4f v = *(const v4f*)(Er + i);
    m = fmaxf(m, fmaxf(fmaxf(v[0], v[1]), fmaxf(v[2], v[3])));
  }
  red[t] = m;
  __syncthreads();
  if ((t & 7) == 0) {
    float mm = red[t];
#pragma unroll
    for (int j = 1; j < 8; ++j) mm = fmaxf(mm, red[t + j]);
    rstat[r] = mm;
  }
  __syncthreads();
  const float rmax = rstat[r];
  float s = 0.f;
  for (int i = 0; i < 256; i += 4) {
    v4f v = *(const v4f*)(Er + i);
    v4f e;
#pragma unroll
    for (int q = 0; q < 4; ++q) e[q] = __expf(v[q] - rmax);
    *(v4f*)(Er + i) = e;
    s += (e[0] + e[1]) + (e[2] + e[3]);
  }
  red[t] = s;
  __syncthreads();
  if ((t & 7) == 0) {
    float ss = red[t];
#pragma unroll
    for (int j = 1; j < 8; ++j) ss += red[t + j];
    rstat[16 + r] = ss;
  }
  __syncthreads();
  const float inv = 1.f / rstat[16 + r];
  float* arow = attn_out + ((size_t)nh * P_ + p0 + r) * P_ + i0;
  __bf16* brow = Abf + (size_t)r * P_ + i0;
  for (int i = 0; i < 256; i += 4) {
    v4f v = *(const v4f*)(Er + i);
    v *= inv;
    __builtin_nontemporal_store(v, (v4f*)(arow + i));
    v4bf b;
#pragma unroll
    for (int q = 0; q < 4; ++q) b[q] = f2bf(v[q]);
    *(v4bf*)(brow + i) = b;
  }
  __syncthreads();

  const int e0 = wave * 16;
  const __bf16* Vb = Vt + ((size_t)nh * HD_ + e0) * P_;
  v8f acc = {};
  for (int s2 = 0; s2 < P_ / 32; ++s2) {
    if (s2 + 1 < P_ / 32) __builtin_prefetch(Vb + (s2 + 1) * 32, 0, 1);
    const __bf16* ap = Abf + (size_t)(lane & 15) * P_ + s2 * 32 + ((lane >> 4) << 3);
    v8bf lo = *(const v8bf*)ap;
    v8bf hi = *(const v8bf*)(ap + 16);
    v16bf a;
#pragma unroll
    for (int q = 0; q < 8; ++q) { a[q] = lo[q]; a[q + 8] = hi[q]; }
    v16bf b = load_ab_bf16(Vb + s2 * 32, P_);
    acc = wmma_bf16(a, b, acc);
  }
#pragma unroll
  for (int j = 0; j < 8; ++j) {
    int p = p0 + mb + j;
    Oh[(size_t)(n * P_ + p) * D_ + h * HD_ + e0 + nc] = f2bf(acc[j]);
  }
}

__global__ void proj_kernel(const __bf16* __restrict__ Oh,
                            const __bf16* __restrict__ Wo_bf,
                            const float* __restrict__ bo,
                            float* __restrict__ out) {
  const int wave = threadIdx.x >> 5;
  const int lane = threadIdx.x & 31;
  const int rg = blockIdx.x >> 5;
  const int ct = blockIdx.x & 31;
  const int r0 = rg * 128 + wave * 16;
  const int e0 = ct * 16;
  const int nc = lane & 15;
  const int mb = (lane >> 4) << 3;

  const __bf16* Ab = Oh + (size_t)r0 * D_;
  const __bf16* Bb = Wo_bf + (size_t)e0 * D_;
  v8f acc = {};
#pragma unroll 4
  for (int s = 0; s < D_ / 32; ++s) {
    v16bf a = load_ab_bf16(Ab + s * 32, D_);
    v16bf b = load_ab_bf16(Bb + s * 32, D_);
    acc = wmma_bf16(a, b, acc);
  }
  const float bias = bo[e0 + nc];
#pragma unroll
  for (int j = 0; j < 8; ++j)
    __builtin_nontemporal_store(acc[j] + bias,
                                out + (size_t)(r0 + mb + j) * D_ + e0 + nc);
}

extern "C" void kernel_launch(void* const* d_in, const int* in_sizes, int n_in,
                              void* d_out, int out_size, void* d_ws, size_t ws_size,
                              hipStream_t stream) {
  const float* x  = (const float*)d_in[0];
  const float* Wq = (const float*)d_in[1];
  const float* Wk = (const float*)d_in[2];
  const float* Wv = (const float*)d_in[3];
  const float* Wo = (const float*)d_in[4];
  const float* bo = (const float*)d_in[5];

  float* out  = (float*)d_out;
  float* attn = out + (size_t)N_ * P_ * D_;

  char* ws = (char*)d_ws;
  __bf16* Q       = (__bf16*)(ws);
  __bf16* K       = (__bf16*)(ws + (((size_t)8) << 20));
  __bf16* Vt      = (__bf16*)(ws + (((size_t)16) << 20));
  __bf16* Oh      = (__bf16*)(ws + (((size_t)24) << 20));
  __bf16* Wqkv_bf = (__bf16*)(ws + (((size_t)32) << 20));
  __bf16* Wo_bf   = (__bf16*)(ws + (((size_t)32) << 20) + 65536);

  const int cvt_groups = (D_ * D_) / 4 + 3 * (HD_ * HD_) / 4;
  cvt_wts_kernel<<<(cvt_groups + 255) / 256, 256, 0, stream>>>(Wq, Wk, Wv, Wo,
                                                               Wqkv_bf, Wo_bf);

  qkv_kernel<<<512, 256, 0, stream>>>(x, Wqkv_bf, Q, K, Vt);

  const size_t smem = (size_t)16 * P_ * 4
                    + (size_t)16 * P_ * 2
                    + (128 + 32) * 4;
  attn_kernel<<<N_ * H_ * (P_ / 16), 128, smem, stream>>>(Q, K, Vt, attn, Oh);

  proj_kernel<<<(N_ * P_ / 128) * (D_ / 16), 256, 0, stream>>>(Oh, Wo_bf, bo, out);
}
